// SO2Block_33105607917793
// MI455X (gfx1250) — compile-verified, matrix operand order checked by reading, not run on hardware
//
#include <hip/hip_runtime.h>

// ============================================================================
// eSCN SO(2) block, fused, for MI455X (gfx1250, wave32, WMMA).
//   - weights pre-converted fp32 -> bf16 into d_ws (L2-resident, ~4MB)
//   - 48 edges per workgroup (3 M-tiles), 8 waves; every B fragment loaded
//     once and reused for 3 M-tiles (3x L2 arithmetic intensity)
//   - v_wmma_f32_16x16x32_bf16 for all GEMMs, f32 accumulate
//   - to_m permutation folded into LDS staging / global scatter
// ============================================================================

#define THREADS 256
#define EPB     48          // edges per block (3 WMMA M-tiles), 30000 = 48*625
#define E_TOT   30000

typedef __attribute__((ext_vector_type(16))) __bf16 v16bf;
typedef __attribute__((ext_vector_type(8)))  __bf16 v8bf;
typedef __attribute__((ext_vector_type(8)))  float  v8f;

// bf16 weight layout in workspace (element offsets)
enum : int {
  OFF_W1D0 = 0,
  OFF_W1M0 = OFF_W1D0 + 256*128,
  OFF_W2M0 = OFF_W1M0 + 256*896,
  OFF_W1D1 = OFF_W2M0 + 896*256,
  OFF_W1R1 = OFF_W1D1 + 512*128,
  OFF_W2R1 = OFF_W1R1 + 256*768,
  OFF_W1I1 = OFF_W2R1 + 768*256,
  OFF_W2I1 = OFF_W1I1 + 256*768,
  OFF_W1D2 = OFF_W2I1 + 768*256,
  OFF_W1R2 = OFF_W1D2 + 512*128,
  OFF_W2R2 = OFF_W1R2 + 256*640,
  OFF_W1I2 = OFF_W2R2 + 640*256,
  OFF_W2I2 = OFF_W1I2 + 256*640,
  W_TOTAL  = OFF_W2I2 + 640*256          // 2,064,384 bf16 = ~4.1 MB
};

// permutation: xp[b] = x[PERM[b]]  and  out[PERM[b]] = y[b]  (to_m orthogonal)
__constant__ int ROWS0[7]  = {0,2,6,11,16,21,26};                      // m=0
__constant__ int ROWS1[12] = {3,7,12,17,22,27,  1,5,10,15,20,25};      // m=1 (r|i)
__constant__ int ROWS2[10] = {8,13,18,23,28,    4,9,14,19,24};         // m=2 (r|i)

// LDS carve (elements, all bf16 now)
#define SXE_N   (EPB*128)     // x_edge bf16                12,288 B
#define SA_N    (EPB*1536)    // A staging bf16 (max m=1)  147,456 B
#define SHID_N  (EPB*1024)    // gated hidden bf16          98,304 B
#define SG_N    (EPB*512)     // silu gates bf16            49,152 B
#define SMEM_BYTES ((SXE_N + SA_N + SHID_N + SG_N)*2)    // 307,200 B <= 320KB

__device__ __forceinline__ unsigned short f2bf(float f) {
  unsigned u = __float_as_uint(f);
  u += 0x7FFFu + ((u >> 16) & 1u);          // round-to-nearest-even
  return (unsigned short)(u >> 16);
}
__device__ __forceinline__ float bf2f(unsigned short h) {
  return __uint_as_float(((unsigned)h) << 16);
}

union BPack { v16bf v; v8bf h[2]; };

// A/B fragment per ISA layout: lane<16 -> K chunks [0..7],[16..23];
// lane>=16 -> [8..15],[24..31]; elements 2i,2i+1 land in VGPR i.
__device__ __forceinline__ v16bf frag16(const unsigned short* rowptr, int k, int half) {
  BPack p;
  p.h[0] = *reinterpret_cast<const v8bf*>(rowptr + k + half*8);
  p.h[1] = *reinterpret_cast<const v8bf*>(rowptr + k + half*8 + 16);
  return p.v;
}

// three 16x16 output tiles (M-halves 0..2), B fragment loaded ONCE per k-step
__device__ __forceinline__ void wmma_mm3(const unsigned short* A, int lda,
                                         const unsigned short* B, int ldb, int K,
                                         v8f& c0, v8f& c1, v8f& c2,
                                         int half, int l15) {
  const unsigned short* a0 = A + (l15     ) * lda;
  const unsigned short* a1 = A + (l15 + 16) * lda;
  const unsigned short* a2 = A + (l15 + 32) * lda;
  const unsigned short* br = B + l15 * ldb;
#pragma unroll 2
  for (int k = 0; k < K; k += 32) {
    v16bf b  = frag16(br, k, half);
    v16bf f0 = frag16(a0, k, half);
    v16bf f1 = frag16(a1, k, half);
    v16bf f2 = frag16(a2, k, half);
    c0 = __builtin_amdgcn_wmma_f32_16x16x32_bf16(false, f0, false, b, (short)0, c0, false, false);
    c1 = __builtin_amdgcn_wmma_f32_16x16x32_bf16(false, f1, false, b, (short)0, c1, false, false);
    c2 = __builtin_amdgcn_wmma_f32_16x16x32_bf16(false, f2, false, b, (short)0, c2, false, false);
  }
}

// stage nrows coefficient rows (permuted) of EPB edges into LDS as bf16
__device__ __forceinline__ void stage_rows(const float* __restrict__ x,
                                           unsigned short* s_A, const int* rows,
                                           int nrows, long e0, int tid) {
  const int per_edge = nrows * 32;                       // float4 chunks / edge
  for (int q = tid; q < EPB * per_edge; q += THREADS) {
    int e   = q / per_edge;
    int rem = q - e * per_edge;
    int r   = rem >> 5;
    int c4  = rem & 31;
    const float4 v = *reinterpret_cast<const float4*>(
        x + (((e0 + e) * 29 + rows[r]) << 7) + c4 * 4);
    unsigned long long p =
        (unsigned long long)f2bf(v.x)        |
        ((unsigned long long)f2bf(v.y) << 16)|
        ((unsigned long long)f2bf(v.z) << 32)|
        ((unsigned long long)f2bf(v.w) << 48);
    *reinterpret_cast<unsigned long long*>(s_A + (size_t)e*(nrows*128) + r*128 + c4*4) = p;
  }
}

// silu(x_edge @ w1d^T + b) -> s_g (bf16, stride 512)
__device__ __forceinline__ void gates(const unsigned short* s_xe, unsigned short* s_g,
                                      const unsigned short* w1d, const float* bias,
                                      int NH, int wv, int half, int l15) {
  for (int ct = wv; ct < (NH >> 4); ct += 8) {
    v8f c0 = (v8f)0.0f, c1 = (v8f)0.0f, c2 = (v8f)0.0f;
    wmma_mm3(s_xe, 128, w1d + ct*16*128, 128, 128, c0, c1, c2, half, l15);
    int col = ct*16 + l15;
    float b = bias[col];
    v8f* cc[3] = {&c0, &c1, &c2};
#pragma unroll
    for (int mh = 0; mh < 3; ++mh)
#pragma unroll
      for (int r = 0; r < 8; ++r) {
        float v = (*cc[mh])[r] + b;
        float g = v * __builtin_amdgcn_rcpf(1.0f + __expf(-v));  // silu, 1 v_rcp
        s_g[(mh*16 + r + half*8)*512 + col] = f2bf(g);
      }
  }
}

// one m>=1 SO2 stage (ms rows real + ms rows imag)
__device__ __forceinline__ void so2_stage(const float* __restrict__ x,
    float* __restrict__ out, const unsigned short* __restrict__ wbf,
    const float* __restrict__ bias, const int* rows, int ms,
    int off_w1d, int off_w1r, int off_w1i, int off_w2r, int off_w2i,
    unsigned short* s_A, unsigned short* s_hid, unsigned short* s_g,
    const unsigned short* s_xe, long e0, int tid, int wv, int half, int l15) {

  const int K1 = ms * 128;
  stage_rows(x, s_A, rows, 2*ms, e0, tid);
  gates(s_xe, s_g, wbf + off_w1d, bias, 512, wv, half, l15);
  __syncthreads();

  // GEMM1: 4 paths {r/i x t0/t1} x 16 hidden col-tiles; gate + bf16 -> s_hid
  for (int tt = wv; tt < 64; tt += 8) {
    int pt = tt >> 4, ct = tt & 15;
    int t = pt & 1, isI = pt >> 1;
    const unsigned short* W1 = wbf + (isI ? off_w1i : off_w1r);
    v8f c0 = (v8f)0.0f, c1 = (v8f)0.0f, c2 = (v8f)0.0f;
    wmma_mm3(s_A + t*K1, 2*K1, W1 + ct*16*K1, K1, K1, c0, c1, c2, half, l15);
    int col = ct*16 + l15;
    v8f* cc[3] = {&c0, &c1, &c2};
#pragma unroll
    for (int mh = 0; mh < 3; ++mh)
#pragma unroll
      for (int r = 0; r < 8; ++r) {
        int m = mh*16 + r + half*8;
        float v = (*cc[mh])[r] * bf2f(s_g[m*512 + isI*256 + col]);
        if (pt == 3) v = -v;                 // xi[t=1] enters out_r with minus
        s_hid[m*1024 + pt*256 + col] = f2bf(v);
      }
  }
  __syncthreads();

  // GEMM2 combined: out_r = xr0 - xi1 ; out_i = xr1 + xi0 (xi1 pre-negated)
  const int tiles_o = ms * 8;
  for (int tt = wv; tt < 2*tiles_o; tt += 8) {
    int o  = tt / tiles_o;
    int ct = tt - o*tiles_o;
    int ptA = o;          // xr term uses hid path t=o
    int ptB = 3 - o;      // xi term: pt3 (t=1,neg) for out_r, pt2 (t=0) for out_i
    v8f c0 = (v8f)0.0f, c1 = (v8f)0.0f, c2 = (v8f)0.0f;
    wmma_mm3(s_hid + ptA*256, 1024, wbf + off_w2r + ct*16*256, 256, 256, c0, c1, c2, half, l15);
    wmma_mm3(s_hid + ptB*256, 1024, wbf + off_w2i + ct*16*256, 256, 256, c0, c1, c2, half, l15);
    int col = ct*16 + l15;
    int rl = col >> 7, cc2 = col & 127;
    int grow = rows[o*ms + rl];
    v8f* cc[3] = {&c0, &c1, &c2};
#pragma unroll
    for (int mh = 0; mh < 3; ++mh)
#pragma unroll
      for (int r = 0; r < 8; ++r)
        out[(((e0 + mh*16 + r + half*8) * 29 + grow) << 7) + cc2] = (*cc[mh])[r];
  }
  __syncthreads();
}

__global__ void __launch_bounds__(THREADS)
so2_fused(const float* __restrict__ x, const float* __restrict__ x_edge,
          const float* __restrict__ b1d0, const float* __restrict__ b1d1,
          const float* __restrict__ b1d2,
          const unsigned short* __restrict__ wbf, float* __restrict__ out) {
  extern __shared__ char smem[];
  unsigned short* s_xe  = (unsigned short*)smem;       // [48][128]  bf16
  unsigned short* s_A   = s_xe  + SXE_N;               // [48][<=1536] bf16
  unsigned short* s_hid = s_A   + SA_N;                // [48][1024] bf16
  unsigned short* s_g   = s_hid + SHID_N;              // [48][512]  bf16

  const int  tid  = threadIdx.x;
  const int  lane = tid & 31;
  const int  wv   = tid >> 5;
  const int  half = lane >> 4;
  const int  l15  = lane & 15;
  const long e0   = (long)blockIdx.x * EPB;

  // stage x_edge -> bf16
  for (int i = tid; i < SXE_N; i += THREADS) {
    int e = i >> 7, c = i & 127;
    s_xe[i] = f2bf(x_edge[((e0 + e) << 7) + c]);
  }
  __syncthreads();

  // ---------------- m = 0 ----------------
  stage_rows(x, s_A, ROWS0, 7, e0, tid);
  gates(s_xe, s_g, wbf + OFF_W1D0, b1d0, 256, wv, half, l15);
  __syncthreads();

  for (int ct = wv; ct < 16; ct += 8) {                 // hidden [48,256]
    v8f c0 = (v8f)0.0f, c1 = (v8f)0.0f, c2 = (v8f)0.0f;
    wmma_mm3(s_A, 896, wbf + OFF_W1M0 + ct*16*896, 896, 896, c0, c1, c2, half, l15);
    int col = ct*16 + l15;
    v8f* cc[3] = {&c0, &c1, &c2};
#pragma unroll
    for (int mh = 0; mh < 3; ++mh)
#pragma unroll
      for (int r = 0; r < 8; ++r) {
        int m = mh*16 + r + half*8;
        s_hid[m*1024 + col] = f2bf((*cc[mh])[r] * bf2f(s_g[m*512 + col]));
      }
  }
  __syncthreads();

  for (int ct = wv; ct < 56; ct += 8) {                 // out0 [48,896]
    v8f c0 = (v8f)0.0f, c1 = (v8f)0.0f, c2 = (v8f)0.0f;
    wmma_mm3(s_hid, 1024, wbf + OFF_W2M0 + ct*16*256, 256, 256, c0, c1, c2, half, l15);
    int col = ct*16 + l15;
    int rl = col >> 7, cc2 = col & 127;
    int grow = ROWS0[rl];
    v8f* cc[3] = {&c0, &c1, &c2};
#pragma unroll
    for (int mh = 0; mh < 3; ++mh)
#pragma unroll
      for (int r = 0; r < 8; ++r)
        out[(((e0 + mh*16 + r + half*8) * 29 + grow) << 7) + cc2] = (*cc[mh])[r];
  }
  __syncthreads();

  // ---------------- m = 1 (ms=6) ----------------
  so2_stage(x, out, wbf, b1d1, ROWS1, 6,
            OFF_W1D1, OFF_W1R1, OFF_W1I1, OFF_W2R1, OFF_W2I1,
            s_A, s_hid, s_g, s_xe, e0, tid, wv, half, l15);

  // ---------------- m = 2 (ms=5) ----------------
  so2_stage(x, out, wbf, b1d2, ROWS2, 5,
            OFF_W1D2, OFF_W1R2, OFF_W1I2, OFF_W2R2, OFF_W2I2,
            s_A, s_hid, s_g, s_xe, e0, tid, wv, half, l15);
}

__global__ void cvt_f32_bf16(const float* __restrict__ src,
                             unsigned short* __restrict__ dst, int n) {
  int i = blockIdx.x * blockDim.x + threadIdx.x;
  if (i < n) dst[i] = f2bf(src[i]);
}

extern "C" void kernel_launch(void* const* d_in, const int* in_sizes, int n_in,
                              void* d_out, int out_size, void* d_ws, size_t ws_size,
                              hipStream_t stream) {
  const float* x      = (const float*)d_in[0];
  const float* x_edge = (const float*)d_in[1];
  const float* w1d0 = (const float*)d_in[2];
  const float* b1d0 = (const float*)d_in[3];
  const float* w1m0 = (const float*)d_in[4];
  const float* w2m0 = (const float*)d_in[5];
  const float* w1d1 = (const float*)d_in[6];
  const float* b1d1 = (const float*)d_in[7];
  const float* w1r1 = (const float*)d_in[8];
  const float* w2r1 = (const float*)d_in[9];
  const float* w1i1 = (const float*)d_in[10];
  const float* w2i1 = (const float*)d_in[11];
  const float* w1d2 = (const float*)d_in[12];
  const float* b1d2 = (const float*)d_in[13];
  const float* w1r2 = (const float*)d_in[14];
  const float* w2r2 = (const float*)d_in[15];
  const float* w1i2 = (const float*)d_in[16];
  const float* w2i2 = (const float*)d_in[17];

  unsigned short* wbf = (unsigned short*)d_ws;

  static bool attr_set = false;   // idempotent host-side attribute (not a stream op)
  if (!attr_set) {
    hipFuncSetAttribute(reinterpret_cast<const void*>(so2_fused),
                        hipFuncAttributeMaxDynamicSharedMemorySize, SMEM_BYTES);
    attr_set = true;
  }

  auto cvt = [&](const float* src, int off, int n) {
    cvt_f32_bf16<<<(n + 255) / 256, 256, 0, stream>>>(src, wbf + off, n);
  };
  cvt(w1d0, OFF_W1D0, 256*128);
  cvt(w1m0, OFF_W1M0, 256*896);
  cvt(w2m0, OFF_W2M0, 896*256);
  cvt(w1d1, OFF_W1D1, 512*128);
  cvt(w1r1, OFF_W1R1, 256*768);
  cvt(w2r1, OFF_W2R1, 768*256);
  cvt(w1i1, OFF_W1I1, 256*768);
  cvt(w2i1, OFF_W2I1, 768*256);
  cvt(w1d2, OFF_W1D2, 512*128);
  cvt(w1r2, OFF_W1R2, 256*640);
  cvt(w2r2, OFF_W2R2, 640*256);
  cvt(w1i2, OFF_W1I2, 256*640);
  cvt(w2i2, OFF_W2I2, 640*256);

  so2_fused<<<E_TOT / EPB, THREADS, SMEM_BYTES, stream>>>(
      x, x_edge, b1d0, b1d1, b1d2, wbf, (float*)d_out);
}
